// TemporalEncoderLayer_63118839382223
// MI455X (gfx1250) — matrix operand order repaired and checked by reading, not compiled, not run on hardware
//
#include <hip/hip_runtime.h>
#include <hip/hip_bf16.h>
#include <math.h>

// ---------------- problem constants ----------------
#define NB   32          // batch
#define FF   4096        // sequence
#define CC   256         // channels
#define HH   8           // heads
#define DD   32          // head dim
#define MTOT (NB * FF)   // 131072 token rows

typedef __attribute__((ext_vector_type(16))) _Float16 v16h;
typedef __attribute__((ext_vector_type(8)))  _Float16 v8h;
typedef __attribute__((ext_vector_type(8)))  float    v8f;
typedef unsigned int u32x4 __attribute__((ext_vector_type(4)));
typedef int          i32x4 __attribute__((ext_vector_type(4)));
typedef int          i32x8 __attribute__((ext_vector_type(8)));

// Compiler-level fence: escapes the LDS pointer and acts as a may-write point so
// the compiler cannot fold LDS loads of TDM-written data to undef.
#define LDS_FENCE(p) asm volatile("" : : "v"((const void*)(p)) : "memory")

// ---------------- WMMA helpers (wave32, 16x16x32 f16) ----------------
__device__ __forceinline__ v8f wmma_f16(v16h a, v16h b, v8f c) {
    return __builtin_amdgcn_wmma_f32_16x16x32_f16(false, a, false, b, (short)0, c, false, false);
}

// A-fragment from row-major f16 (lda elems): lane L: M=L&15, K = kb..kb+15, kb=(L>>4)*16
__device__ __forceinline__ v16h load_a_frag(const _Float16* src, int lda) {
    const int lane = threadIdx.x & 31;
    const _Float16* p = src + (size_t)(lane & 15) * lda + ((lane >> 4) * 16);
    v8h lo = *reinterpret_cast<const v8h*>(p);
    v8h hi = *reinterpret_cast<const v8h*>(p + 8);
    return __builtin_shufflevector(lo, hi, 0,1,2,3,4,5,6,7,8,9,10,11,12,13,14,15);
}

__device__ __forceinline__ float elu1f(float x) { return x > 0.0f ? x + 1.0f : __expf(x); }
__device__ __forceinline__ float gelu_exact(float x) {
    return 0.5f * x * (1.0f + erff(x * 0.70710678118654752f));
}

// ---------------- Tensor Data Mover: 2D f16 tile -> LDS ----------------
// D# per CDNA5 ISA: group0 {count=1, lds_addr, global_addr, type=2},
// group1 {data_size=2B, tensor_dim0/1 = tile dims (no OOB), tile_dim0/1, dim0_stride}.
__device__ __forceinline__ void tdm_load_2d(const _Float16* gsrc, unsigned lds_byte_off,
                                            unsigned kelems, unsigned rows, unsigned stride_elems) {
    unsigned long long ga = (unsigned long long)(uintptr_t)gsrc;
    u32x4 g0;
    g0[0] = 1u;                                                  // count=1
    g0[1] = lds_byte_off;                                        // lds_addr
    g0[2] = (unsigned)(ga & 0xFFFFFFFFu);                        // global_addr lo
    g0[3] = (unsigned)((ga >> 32) & 0x1FFFFFFu) | (2u << 30);    // global_addr hi + type=2
    i32x8 g1;
    g1[0] = (int)(1u << 16);                                     // data_size=1 -> 2 bytes
    g1[1] = (int)((kelems & 0xFFFFu) << 16);                     // tensor_dim0 lo16 in [31:16]
    g1[2] = (int)((kelems >> 16) & 0xFFFFu) | (int)((rows & 0xFFFFu) << 16); // dim0 hi | dim1 lo
    g1[3] = (int)((rows >> 16) & 0xFFFFu) | (int)((kelems & 0xFFFFu) << 16); // dim1 hi | tile_dim0
    g1[4] = (int)(rows & 0xFFFFu);                               // tile_dim1 (tile_dim2=0)
    g1[5] = (int)stride_elems;                                   // tensor_dim0_stride lo32
    g1[6] = 0;
    g1[7] = 0;
    i32x4 z4 = {0, 0, 0, 0};
#if __clang_major__ >= 23
    i32x8 z8 = {0, 0, 0, 0, 0, 0, 0, 0};
    __builtin_amdgcn_tensor_load_to_lds(g0, g1, z4, z4, z8, 0);
#else
    __builtin_amdgcn_tensor_load_to_lds(g0, g1, z4, z4, 0);
#endif
}

// ---------------- fp32 -> fp16 convert ----------------
__global__ void cvt_f32_to_f16(const float* __restrict__ src, _Float16* __restrict__ dst, size_t n) {
    size_t i = (size_t)blockIdx.x * blockDim.x + threadIdx.x;
    size_t stride = (size_t)gridDim.x * blockDim.x;
    for (; i < n; i += stride) dst[i] = (_Float16)src[i];
}

// ---------------- TDM double-buffered GEMM core ----------------
// Block = 256 thr (8 waves). Block tile: 128 rows x 128 cols, K stepped by 32.
// LDS: A[2][128][32] f16 (16KB) + B[2][128][32] f16 (16KB). Wave0 drives the TDM pipeline.
#define ABUF (128 * 32)
#define BBUF (128 * 32)
#define NT   8            // 16-wide col tiles per wave

__device__ __forceinline__ void gemm_tdm_core(const _Float16* __restrict__ aglob,
                                              const _Float16* __restrict__ wglob,
                                              _Float16* smem, v8f acc[NT]) {
    const int wave = threadIdx.x >> 5;
    const int lane = threadIdx.x & 31;
    const int rowblk = blockIdx.x * 128;
    const int col0   = blockIdx.y * 128;

    if (wave == 0) {  // prologue: stage K-step 0 into buffer 0
        tdm_load_2d(aglob + (size_t)rowblk * CC, 0u, 32, 128, CC);
        tdm_load_2d(wglob + (size_t)col0 * CC, (unsigned)(2 * ABUF * sizeof(_Float16)), 32, 128, CC);
    }
#pragma unroll
    for (int s = 0; s < CC / 32; ++s) {
        const int cur = s & 1;
        if (wave == 0) __builtin_amdgcn_s_wait_tensorcnt(0);
        __syncthreads();                                   // LDS buffer `cur` now complete
        LDS_FENCE(smem);                                   // may-write point: keep the ds loads
        if (wave == 0 && s + 1 < CC / 32) {                // prefetch next K-step into other buffer
            const int nk = (s + 1) * 32;
            tdm_load_2d(aglob + (size_t)rowblk * CC + nk,
                        (unsigned)(((cur ^ 1) * ABUF) * sizeof(_Float16)), 32, 128, CC);
            tdm_load_2d(wglob + (size_t)col0 * CC + nk,
                        (unsigned)((2 * ABUF + (cur ^ 1) * BBUF) * sizeof(_Float16)), 32, 128, CC);
        }
        const _Float16* As = smem + cur * ABUF;            // [128][32]
        const _Float16* Bs = smem + 2 * ABUF + cur * BBUF; // [128][32]
        // Issue ALL fragment loads first, then the WMMA chain: one dscnt wait
        // covers the whole chain instead of a wait per WMMA.
        v16h a = load_a_frag(As + (size_t)(wave * 16) * 32, 32);
        v16h b[NT];
#pragma unroll
        for (int t = 0; t < NT; ++t)
            b[t] = *reinterpret_cast<const v16h*>(
                Bs + (size_t)(16 * t + (lane & 15)) * 32 + ((lane >> 4) * 16));
#pragma unroll
        for (int t = 0; t < NT; ++t)
            acc[t] = wmma_f16(a, b[t], acc[t]);
        __syncthreads();                                   // all waves done with `cur` before overwrite
    }
}

// ---------------- QKV GEMM + elu+1 feature map + scatter to [N,H,F,D] ----------------
__global__ __launch_bounds__(256) void gemm_qkv_kernel(
        const _Float16* __restrict__ xh, const _Float16* __restrict__ wq,
        _Float16* __restrict__ qo, _Float16* __restrict__ ko, _Float16* __restrict__ vo) {
    __shared__ _Float16 smem[2 * ABUF + 2 * BBUF];
    const int wave = threadIdx.x >> 5;
    const int lane = threadIdx.x & 31;
    v8f acc[NT] = {};
    gemm_tdm_core(xh, wq, smem, acc);

    const int row0 = blockIdx.x * 128 + wave * 16;
    const int col0 = blockIdx.y * 128;
#pragma unroll
    for (int t = 0; t < NT; ++t) {
        const int o  = col0 + 16 * t + (lane & 15);
        const int wh = o >> 8;
        const int hd = o & 255;
        const int h  = hd >> 5;
        const int d  = hd & 31;
#pragma unroll
        for (int i = 0; i < 8; ++i) {
            const int m = row0 + i + 8 * (lane >> 4);
            const int n = m >> 12;
            const int f = m & (FF - 1);
            const size_t dst = (((size_t)(n * HH + h)) * FF + f) * DD + d;
            float val = acc[t][i];
            if (wh == 0)      qo[dst] = (_Float16)elu1f(val);
            else if (wh == 1) ko[dst] = (_Float16)elu1f(val);
            else              vo[dst] = (_Float16)val;
        }
    }
}

// ---------------- FFN GEMM1: h = gelu(x1h @ W1^T + b1) -> f16 ----------------
__global__ __launch_bounds__(256) void gemm_gelu_kernel(
        const _Float16* __restrict__ xh, const _Float16* __restrict__ w,
        const float* __restrict__ bias, _Float16* __restrict__ out) {
    __shared__ _Float16 smem[2 * ABUF + 2 * BBUF];
    const int wave = threadIdx.x >> 5;
    const int lane = threadIdx.x & 31;
    v8f acc[NT] = {};
    gemm_tdm_core(xh, w, smem, acc);

    const int row0 = blockIdx.x * 128 + wave * 16;
    const int col0 = blockIdx.y * 128;
#pragma unroll
    for (int t = 0; t < NT; ++t) {
        const int o = col0 + 16 * t + (lane & 15);
        const float bo = bias[o];
#pragma unroll
        for (int i = 0; i < 8; ++i) {
            const int m = row0 + i + 8 * (lane >> 4);
            out[(size_t)m * CC + o] = (_Float16)gelu_exact(acc[t][i] + bo);
        }
    }
}

// ---------------- FFN GEMM2: y = h @ W2^T + b2 -> f32 ----------------
__global__ __launch_bounds__(256) void gemm_bias_kernel(
        const _Float16* __restrict__ xh, const _Float16* __restrict__ w,
        const float* __restrict__ bias, float* __restrict__ out) {
    __shared__ _Float16 smem[2 * ABUF + 2 * BBUF];
    const int wave = threadIdx.x >> 5;
    const int lane = threadIdx.x & 31;
    v8f acc[NT] = {};
    gemm_tdm_core(xh, w, smem, acc);

    const int row0 = blockIdx.x * 128 + wave * 16;
    const int col0 = blockIdx.y * 128;
#pragma unroll
    for (int t = 0; t < NT; ++t) {
        const int o = col0 + 16 * t + (lane & 15);
        const float bo = bias[o];
#pragma unroll
        for (int i = 0; i < 8; ++i) {
            const int m = row0 + i + 8 * (lane >> 4);
            out[(size_t)m * CC + o] = acc[t][i] + bo;
        }
    }
}

// ---------------- kv = k^T v (32x32 per (n,h)) + ksum ----------------
__global__ __launch_bounds__(256) void kv_ksum_kernel(
        const _Float16* __restrict__ kh, const _Float16* __restrict__ vh,
        _Float16* __restrict__ kvh, float* __restrict__ ksum) {
    const int nh   = blockIdx.x;
    const int tid  = threadIdx.x;
    const int wave = tid >> 5;
    const int lane = tid & 31;
    const int tile = wave & 3, trow = tile >> 1, tcol = tile & 1;
    const int khalf = wave >> 2;
    const _Float16* kb_ = kh + (size_t)nh * FF * DD;
    const _Float16* vb_ = vh + (size_t)nh * FF * DD;

    __shared__ float red[8][32][8];
    __shared__ float ksl[256];

    v8f acc = v8f{};
    const int d1 = (lane & 15) + 16 * trow;
    const int d2 = (lane & 15) + 16 * tcol;
    const int fbase = khalf * (FF / 2);
    for (int f0 = fbase; f0 < fbase + FF / 2; f0 += 32) {
        v16h a, b;
        const int fa = f0 + (lane >> 4) * 16;
        const int fb = f0 + (lane >> 4) * 16;
#pragma unroll
        for (int i = 0; i < 8; ++i) {
            a[i]     = kb_[(size_t)(fa + i)      * DD + d1];
            a[i + 8] = kb_[(size_t)(fa + i + 8) * DD + d1];
        }
#pragma unroll
        for (int i = 0; i < 16; ++i)
            b[i] = vb_[(size_t)(fb + i) * DD + d2];
        acc = wmma_f16(a, b, acc);
    }
#pragma unroll
    for (int i = 0; i < 8; ++i) red[wave][lane][i] = acc[i];
    {
        const int d = tid & 31;
        float p = 0.0f;
        for (int f = tid >> 5; f < FF; f += 8) p += (float)kb_[(size_t)f * DD + d];
        ksl[tid] = p;
    }
    __syncthreads();

    if (wave < 4) {
#pragma unroll
        for (int i = 0; i < 8; ++i) {
            float s = red[wave][lane][i] + red[wave + 4][lane][i];
            const int r = i + 8 * (lane >> 4) + 16 * trow;
            const int c = (lane & 15) + 16 * tcol;
            kvh[(size_t)nh * (DD * DD) + (size_t)c * DD + r] = (_Float16)s;
        }
    }
    if (tid < 32) {
        float s = 0.0f;
#pragma unroll
        for (int j = 0; j < 8; ++j) s += ksl[j * 32 + tid];
        ksum[(size_t)nh * DD + tid] = s;
    }
}

// ---------------- attention out: t[l,m] = z_l * sum_d q[l,d] kv[m,d] ----------------
__global__ __launch_bounds__(256) void attn_out_kernel(
        const _Float16* __restrict__ qh, const _Float16* __restrict__ kvh,
        const float* __restrict__ ksum, float* __restrict__ tout) {
    const int wave = threadIdx.x >> 5;
    const int lane = threadIdx.x & 31;
    const int nh   = blockIdx.y;
    const int f0   = blockIdx.x * 128 + wave * 16;
    const int n    = nh >> 3, h = nh & 7;
    const _Float16* qb = qh + (size_t)nh * FF * DD;

    __shared__ float zl[8][16];
    if (lane < 16) {
        const int l = f0 + lane;
        float s = 0.0f;
#pragma unroll
        for (int d = 0; d < DD; ++d)
            s += (float)qb[(size_t)l * DD + d] * ksum[(size_t)nh * DD + d];
        zl[wave][lane] = 1.0f / (s + 1e-6f);
    }
    __syncthreads();

    v16h a = load_a_frag(qb + (size_t)f0 * DD, DD);
#pragma unroll
    for (int t = 0; t < 2; ++t) {
        v16h b = *reinterpret_cast<const v16h*>(
            kvh + (size_t)nh * (DD * DD) + (size_t)((lane & 15) + 16 * t) * DD + (lane >> 4) * 16);
        v8f acc = wmma_f16(a, b, v8f{});
        const int m = (lane & 15) + 16 * t;
#pragma unroll
        for (int i = 0; i < 8; ++i) {
            const int ri = i + 8 * (lane >> 4);
            tout[((size_t)n * FF + (f0 + ri)) * CC + h * DD + m] = acc[i] * zl[wave][ri];
        }
    }
}

// ---------------- AddNorm: out = LN(a + b) * g + beta ; optional f16 copy ----------------
__global__ __launch_bounds__(256) void addnorm_kernel(
        const float* __restrict__ a, const float* __restrict__ b,
        const float* __restrict__ g, const float* __restrict__ beta,
        float* __restrict__ out32, _Float16* __restrict__ out16) {
    const int row = blockIdx.x;
    const int tid = threadIdx.x;
    const size_t idx = (size_t)row * CC + tid;
    float v = a[idx] + b[idx];

    __shared__ float s1[256], s2[256];
    s1[tid] = v; s2[tid] = v * v;
    __syncthreads();
#pragma unroll
    for (int off = 128; off > 0; off >>= 1) {
        if (tid < off) { s1[tid] += s1[tid + off]; s2[tid] += s2[tid + off]; }
        __syncthreads();
    }
    const float mu  = s1[0] * (1.0f / CC);
    const float var = s2[0] * (1.0f / CC) - mu * mu;
    const float y = (v - mu) * rsqrtf(var + 1e-5f) * g[tid] + beta[tid];
    out32[idx] = y;
    if (out16) out16[idx] = (_Float16)y;
}

// ---------------- launcher ----------------
extern "C" void kernel_launch(void* const* d_in, const int* in_sizes, int n_in,
                              void* d_out, int out_size, void* d_ws, size_t ws_size,
                              hipStream_t stream) {
    const float* x     = (const float*)d_in[0];
    const float* Wqkv  = (const float*)d_in[1];
    const float* ln1_g = (const float*)d_in[2];
    const float* ln1_b = (const float*)d_in[3];
    const float* W1    = (const float*)d_in[4];
    const float* b1    = (const float*)d_in[5];
    const float* W2    = (const float*)d_in[6];
    const float* b2    = (const float*)d_in[7];
    const float* ln2_g = (const float*)d_in[8];
    const float* ln2_b = (const float*)d_in[9];
    float* out = (float*)d_out;

    char* ws = (char*)d_ws;
    const size_t MB = 1ull << 20;
    _Float16* xh    = (_Float16*)(ws + 0 * MB);       // 64 MB  (reused as h)
    _Float16* qh    = (_Float16*)(ws + 64 * MB);      // 64 MB  (reused as x1h)
    _Float16* kh    = (_Float16*)(ws + 128 * MB);     // 64 MB
    _Float16* vh    = (_Float16*)(ws + 192 * MB);     // 64 MB
    _Float16* kvh   = (_Float16*)(ws + 256 * MB);     // 512 KB
    float*    ksum  = (float*)   (ws + 257 * MB);     // 32 KB
    float*    tbuf  = (float*)   (ws + 258 * MB);     // 128 MB (reused as y)
    float*    x1    = (float*)   (ws + 386 * MB);     // 128 MB
    _Float16* wqkvh = (_Float16*)(ws + 514 * MB);     // 384 KB
    _Float16* w1h   = (_Float16*)(ws + 515 * MB);     // 128 KB
    _Float16* w2h   = (_Float16*)(ws + 516 * MB);     // 128 KB
    _Float16* x1h   = qh;
    _Float16* hh    = xh;
    float*    ybuf  = tbuf;

    cvt_f32_to_f16<<<2048, 256, 0, stream>>>(x, xh, (size_t)MTOT * CC);
    cvt_f32_to_f16<<<192, 256, 0, stream>>>(Wqkv, wqkvh, (size_t)3 * CC * CC);
    cvt_f32_to_f16<<<64, 256, 0, stream>>>(W1, w1h, (size_t)CC * CC);
    cvt_f32_to_f16<<<64, 256, 0, stream>>>(W2, w2h, (size_t)CC * CC);

    gemm_qkv_kernel<<<dim3(MTOT / 128, (3 * CC) / 128), 256, 0, stream>>>(xh, wqkvh, qh, kh, vh);
    kv_ksum_kernel<<<NB * HH, 256, 0, stream>>>(kh, vh, kvh, ksum);
    attn_out_kernel<<<dim3(FF / 128, NB * HH), 256, 0, stream>>>(qh, kvh, ksum, tbuf);
    addnorm_kernel<<<MTOT, 256, 0, stream>>>(tbuf, x, ln1_g, ln1_b, x1, x1h);
    gemm_gelu_kernel<<<dim3(MTOT / 128, CC / 128), 256, 0, stream>>>(x1h, w1h, b1, hh);
    gemm_bias_kernel<<<dim3(MTOT / 128, CC / 128), 256, 0, stream>>>(hh, w2h, b2, ybuf);
    addnorm_kernel<<<MTOT, 256, 0, stream>>>(ybuf, x1, ln2_g, ln2_b, out, nullptr);
}
